// atten_layer_70128226009232
// MI455X (gfx1250) — compile-verified, matrix operand order checked by reading, not run on hardware
//
#include <hip/hip_runtime.h>

typedef __attribute__((ext_vector_type(16))) _Float16 v16h;
typedef __attribute__((ext_vector_type(2)))  _Float16 v2h;
typedef __attribute__((ext_vector_type(8)))  float    v8f;

constexpr int Sdim  = 2048;
constexpr int Ddim  = 128;
constexpr int QTILE = 128;   // query rows per block (8 waves x 16)
constexpr int KTILE = 64;    // keys per iteration (4 x 16x16 score tiles)
constexpr int NWAVES = 8;

// element e of a 16-bit A/B fragment -> contraction-dim offset (ISA 7.12.2)
#define KOFF(e) ((( ((e) >> 1) < 4 ) ? (((e) >> 1) * 2) : (16 + (((e) >> 1) - 4) * 2)) \
                 + (hi << 3) + ((e) & 1))

// DPP16 row_ror:n — VALU cross-lane rotate within each 16-lane row (no LDS traffic)
#define DPP_ROR(x, n) __builtin_bit_cast(float, __builtin_amdgcn_update_dpp(      \
    0, __builtin_bit_cast(int, (x)), 0x120 + (n), 0xf, 0xf, true))

__device__ __forceinline__ float rowmax16(float x) {
    x = fmaxf(x, DPP_ROR(x, 8));
    x = fmaxf(x, DPP_ROR(x, 4));
    x = fmaxf(x, DPP_ROR(x, 2));
    x = fmaxf(x, DPP_ROR(x, 1));
    return x;
}
__device__ __forceinline__ float rowsum16(float x) {
    x += DPP_ROR(x, 8);
    x += DPP_ROR(x, 4);
    x += DPP_ROR(x, 2);
    x += DPP_ROR(x, 1);
    return x;
}

__global__ __launch_bounds__(256) void fattn_kernel(
    const float* __restrict__ Q, const float* __restrict__ K,
    const float* __restrict__ V, float* __restrict__ O)
{
    __shared__ _Float16 Kl[KTILE * Ddim];          // K tile, [k][d], f16      (16 KB)
    __shared__ _Float16 Vt[Ddim * KTILE];          // V tile, [d][k] transposed (16 KB)
    __shared__ _Float16 Pl[NWAVES * 16 * KTILE];   // per-wave P relayout       (16 KB)

    const int t    = threadIdx.x;
    const int wave = t >> 5;
    const int lane = t & 31;
    const int hi   = lane >> 4;   // half-wave select
    const int ln   = lane & 15;

    const int bh = blockIdx.y;                       // 0..63  (B*H)
    const int qt = blockIdx.x;                       // 0..15  (S/QTILE)
    const size_t base  = (size_t)bh * Sdim * Ddim;
    const int    qrow0 = qt * QTILE + wave * 16;

    const float scale = 0.08838834764831845f;        // 1/sqrt(128)

    // ---- persistent Q A-fragments (4 x 16x32 f16), pre-scaled ----
    v16h qf[4];
    {
        const float* qrow = Q + base + (size_t)(qrow0 + ln) * Ddim;
#pragma unroll
        for (int s = 0; s < 4; ++s) {
#pragma unroll
            for (int e = 0; e < 16; ++e)
                qf[s][e] = (_Float16)(qrow[s * 32 + KOFF(e)] * scale);
        }
    }

    // ---- fp32 accumulators: 16 rows x 128 cols as 8 C-fragments ----
    v8f acc[8];
#pragma unroll
    for (int c = 0; c < 8; ++c) acc[c] = {};
    float mrow[8], lrow[8];
#pragma unroll
    for (int r = 0; r < 8; ++r) { mrow[r] = -__builtin_inff(); lrow[r] = 0.0f; }

    // cooperative-load mapping: each thread owns 2 consecutive key rows x 16 dims
    const int kr2  = (t >> 3) * 2;   // 0,2,..,62
    const int dblk = (t & 7) * 16;   // 0..112

    for (int kb0 = 0; kb0 < Sdim; kb0 += KTILE) {
        __syncthreads();  // previous tile fully consumed
        {
            const float* kp0 = K + base + (size_t)(kb0 + kr2)     * Ddim + dblk;
            const float* kp1 = K + base + (size_t)(kb0 + kr2 + 1) * Ddim + dblk;
            const float* vp0 = V + base + (size_t)(kb0 + kr2)     * Ddim + dblk;
            const float* vp1 = V + base + (size_t)(kb0 + kr2 + 1) * Ddim + dblk;
#pragma unroll
            for (int i = 0; i < 4; ++i) {
                float4 k0 = *reinterpret_cast<const float4*>(kp0 + i * 4);
                float4 k1 = *reinterpret_cast<const float4*>(kp1 + i * 4);
                float4 w0 = *reinterpret_cast<const float4*>(vp0 + i * 4);
                float4 w1 = *reinterpret_cast<const float4*>(vp1 + i * 4);
                const int d0 = dblk + i * 4;
                // K tile: row-major halves (consecutive -> merges to ds_store_b128)
                *reinterpret_cast<v2h*>(&Kl[(kr2)     * Ddim + d0])     = (v2h){(_Float16)k0.x, (_Float16)k0.y};
                *reinterpret_cast<v2h*>(&Kl[(kr2)     * Ddim + d0 + 2]) = (v2h){(_Float16)k0.z, (_Float16)k0.w};
                *reinterpret_cast<v2h*>(&Kl[(kr2 + 1) * Ddim + d0])     = (v2h){(_Float16)k1.x, (_Float16)k1.y};
                *reinterpret_cast<v2h*>(&Kl[(kr2 + 1) * Ddim + d0 + 2]) = (v2h){(_Float16)k1.z, (_Float16)k1.w};
                // V tile transposed: pack the two key rows -> single b32 store per dim
                *reinterpret_cast<v2h*>(&Vt[(d0 + 0) * KTILE + kr2]) = (v2h){(_Float16)w0.x, (_Float16)w1.x};
                *reinterpret_cast<v2h*>(&Vt[(d0 + 1) * KTILE + kr2]) = (v2h){(_Float16)w0.y, (_Float16)w1.y};
                *reinterpret_cast<v2h*>(&Vt[(d0 + 2) * KTILE + kr2]) = (v2h){(_Float16)w0.z, (_Float16)w1.z};
                *reinterpret_cast<v2h*>(&Vt[(d0 + 3) * KTILE + kr2]) = (v2h){(_Float16)w0.w, (_Float16)w1.w};
            }
        }
        __syncthreads();

        // hide next tile's HBM latency behind this tile's WMMA work
        if (kb0 + KTILE < Sdim) {
            __builtin_prefetch(K + base + (size_t)(kb0 + KTILE + kr2) * Ddim + dblk, 0, 0);
            __builtin_prefetch(K + base + (size_t)(kb0 + KTILE + kr2 + 1) * Ddim + dblk, 0, 0);
            __builtin_prefetch(V + base + (size_t)(kb0 + KTILE + kr2) * Ddim + dblk, 0, 0);
            __builtin_prefetch(V + base + (size_t)(kb0 + KTILE + kr2 + 1) * Ddim + dblk, 0, 0);
        }

        // ---- scores: S = Q x K^T for four 16-key tiles, K-dim 128 chained ----
        v8f sc[4];
#pragma unroll
        for (int j = 0; j < 4; ++j) sc[j] = {};
#pragma unroll
        for (int s = 0; s < 4; ++s) {
#pragma unroll
            for (int j = 0; j < 4; ++j) {
                v16h b;
#pragma unroll
                for (int e = 0; e < 16; ++e)
                    b[e] = Kl[(j * 16 + ln) * Ddim + s * 32 + KOFF(e)];
                sc[j] = __builtin_amdgcn_wmma_f32_16x16x32_f16(false, qf[s], false, b,
                                                               (short)0, sc[j], false, false);
            }
        }

        // ---- online softmax: VALU-only DPP butterflies, no LDS round-trips ----
        _Float16* Pw = Pl + wave * 16 * KTILE;
#pragma unroll
        for (int r = 0; r < 8; ++r) {
            float v0 = sc[0][r], v1 = sc[1][r], v2 = sc[2][r], v3 = sc[3][r];
            float m  = rowmax16(fmaxf(fmaxf(v0, v1), fmaxf(v2, v3)));
            const float mn    = fmaxf(mrow[r], m);
            const float alpha = __expf(mrow[r] - mn);
            mrow[r] = mn;
            const float p0 = __expf(v0 - mn);
            const float p1 = __expf(v1 - mn);
            const float p2 = __expf(v2 - mn);
            const float p3 = __expf(v3 - mn);
            lrow[r] = lrow[r] * alpha + rowsum16(p0 + p1 + p2 + p3);
#pragma unroll
            for (int c = 0; c < 8; ++c) acc[c][r] *= alpha;
            const int row = r + 8 * hi;
            Pw[row * KTILE +      ln] = (_Float16)p0;
            Pw[row * KTILE + 16 + ln] = (_Float16)p1;
            Pw[row * KTILE + 32 + ln] = (_Float16)p2;
            Pw[row * KTILE + 48 + ln] = (_Float16)p3;
        }

        // ---- relayout P (C-layout -> two A-fragments, k=0..31 and 32..63) ----
        v16h pf0, pf1;
#pragma unroll
        for (int e = 0; e < 16; ++e) {
            pf0[e] = Pw[ln * KTILE +      KOFF(e)];
            pf1[e] = Pw[ln * KTILE + 32 + KOFF(e)];
        }

        // ---- O += P x V : 8 N-chunks of 16 dims, contraction K = 64 keys ----
#pragma unroll
        for (int c = 0; c < 8; ++c) {
            v16h vf0, vf1;
#pragma unroll
            for (int e = 0; e < 16; ++e) {
                vf0[e] = Vt[(c * 16 + ln) * KTILE +      KOFF(e)];
                vf1[e] = Vt[(c * 16 + ln) * KTILE + 32 + KOFF(e)];
            }
            acc[c] = __builtin_amdgcn_wmma_f32_16x16x32_f16(false, pf0, false, vf0,
                                                            (short)0, acc[c], false, false);
            acc[c] = __builtin_amdgcn_wmma_f32_16x16x32_f16(false, pf1, false, vf1,
                                                            (short)0, acc[c], false, false);
        }
    }

    // ---- epilogue: divide by softmax denominator, store fp32 ----
#pragma unroll
    for (int r = 0; r < 8; ++r) {
        const float inv = 1.0f / lrow[r];
        float* orow = O + base + (size_t)(qrow0 + r + 8 * hi) * Ddim;
#pragma unroll
        for (int c = 0; c < 8; ++c)
            orow[c * 16 + ln] = acc[c][r] * inv;
    }
}

extern "C" void kernel_launch(void* const* d_in, const int* in_sizes, int n_in,
                              void* d_out, int out_size, void* d_ws, size_t ws_size,
                              hipStream_t stream) {
    (void)in_sizes; (void)n_in; (void)out_size; (void)d_ws; (void)ws_size;
    const float* Q = (const float*)d_in[0];
    const float* K = (const float*)d_in[1];
    const float* V = (const float*)d_in[2];
    float*       O = (float*)d_out;

    dim3 grid(Sdim / QTILE, 4 * 16);   // (16 q-tiles, B*H = 64)
    dim3 block(256);                   // 8 wave32 waves
    fattn_kernel<<<grid, block, 0, stream>>>(Q, K, V, O);
}
